// Model_37649683316771
// MI455X (gfx1250) — compile-verified, matrix operand order checked by reading, not run on hardware
//
#include <hip/hip_runtime.h>
#include <hip/hip_fp16.h>

// ---------------------------------------------------------------------------
// RNNImputation + OS-CNN head for MI455X (gfx1250, wave32, WMMA)
//   out[16]   = fc_out( GAP( conv-head( new_input ) ) )
//   pred[4784]= fc_imp( h_0..h_298 )
// d_out = [ out (16) | prediction (16*299) ]  (f32)
// ---------------------------------------------------------------------------

typedef __attribute__((ext_vector_type(16))) _Float16 v16h;
typedef __attribute__((ext_vector_type(8)))  float    v8f;

constexpr int NB   = 16;        // batch
constexpr int NT   = 300;       // time
constexpr int NH   = 128;       // hidden
constexpr int NG   = 384;       // 3*NH
constexpr int NPOS = NB * NT;   // 4800
constexpr int NC1  = 9636;      // layer-1 channels (6 * 1606)
constexpr int C1G  = 1606;
constexpr int NC3  = 28908;     // layer-3 channels (3 * 9636)
constexpr int C3G  = 9636;
constexpr float FEPS = 1e-5f;
constexpr float SENT = 128.0f;

// primes = 7,11,15,19,23,27 == 7+4*i
__device__ __forceinline__ int prime_k(int i) { return 7 + 4 * i; }

// Native-rate nonlinearities: v_exp_f32 + v_rcp_f32 (single trans ops) instead
// of the IEEE division expansion — this sits on the serial GRU critical path.
__device__ __forceinline__ float fast_sigmoid(float x) {
  return __builtin_amdgcn_rcpf(1.f + __expf(-x));
}
__device__ __forceinline__ float fast_tanh(float x) {
  // tanh(x) = 1 - 2/(exp(2x)+1)
  return 1.f - 2.f * __builtin_amdgcn_rcpf(1.f + __expf(2.f * x));
}

// dilated (rate 4) SAME conv over one 300-sample row, in_ch = 1
__device__ __forceinline__ float conv1_eval(const float* __restrict__ w, int k,
                                            const float* xb, int t) {
  float y = 0.f;
  const int half = (k - 1) >> 1;
  for (int j = 0; j < k; ++j) {
    const int idx = t + (j - half) * 4;
    if (idx >= 0 && idx < NT) y += w[j] * xb[idx];
  }
  return y;
}

// two-value block reduction (256 threads)
__device__ __forceinline__ void blk_reduce2(float& a, float& b, float* red, int tid) {
  __syncthreads();
  red[tid] = a; red[256 + tid] = b;
  __syncthreads();
  for (int off = 128; off > 0; off >>= 1) {
    if (tid < off) { red[tid] += red[tid + off]; red[256 + tid] += red[256 + tid + off]; }
    __syncthreads();
  }
  a = red[0]; b = red[256];
}

// ===========================================================================
// Kernel 1: GRU recurrence, fused imputation + prediction + conv-input stream
//   1 block x 256 threads (8 wave32).  gh = h @ w_hh.T via WMMA f16->f32:
//   24 N-tiles of 16, K=128 -> 4 x v_wmma_f32_16x16x32_f16 per tile.
//   w_hh B-fragments packed once into VGPRs and reused for all 300 steps.
// ===========================================================================
__global__ __launch_bounds__(256, 1)
void k_gru(const float* __restrict__ x_in,
           const float* __restrict__ w_ih, const float* __restrict__ w_hh,
           const float* __restrict__ b_ih, const float* __restrict__ b_hh,
           const float* __restrict__ fci_w, const float* __restrict__ fci_b,
           float* __restrict__ pred,   // [NB*(NT-1)]
           float* __restrict__ xin)    // ws: conv-head input [NB*NT]
{
  __shared__ float    h_s  [NB][NH];
  __shared__ _Float16 h16_s[NB][NH];
  __shared__ float    gh_s [NB][NG];
  __shared__ float    cur_s[NB];
  __shared__ float    pr_s [NB];

  const int tid  = threadIdx.x;
  const int lane = tid & 31;
  const int wv   = tid >> 5;     // wave 0..7 -> N-tiles wv*3 .. wv*3+2
  const int n    = lane & 15;    // column within 16x16 tile
  const int hb   = lane >> 4;    // lane-half (K sub-block select)

  // ---- pack w_hh into WMMA B fragments (held in VGPRs across all steps) ----
  // B (32x16) layout assumption (mirror of documented 16-bit A layout):
  //   lanes 0-15 : N=lane,    K = kb*32 + {0..7, 16..23}
  //   lanes16-31 : N=lane-16, K = kb*32 + {8..15, 24..31}
  v16h Bf[3][4];
  float binit[3];
#pragma unroll
  for (int tt = 0; tt < 3; ++tt) {
    const int g = (wv * 3 + tt) * 16 + n;   // gate column in [0,384)
    binit[tt] = b_hh[g];
#pragma unroll
    for (int kb = 0; kb < 4; ++kb) {
      v16h f;
#pragma unroll
      for (int i = 0; i < 8; ++i) {
        f[i]     = (_Float16)w_hh[g * NH + kb * 32 + hb * 8 + i];
        f[8 + i] = (_Float16)w_hh[g * NH + kb * 32 + 16 + hb * 8 + i];
      }
      Bf[tt][kb] = f;
    }
  }

  // ---- per-thread gate constants: each thread owns column jj for 8 batches
  const int jj  = tid & 127;
  const int bb0 = tid >> 7;                         // element b = bb0 + 2*i
  const float wir = w_ih[jj], wiz = w_ih[NH + jj], win = w_ih[2 * NH + jj];
  const float bir = b_ih[jj], biz = b_ih[NH + jj], bin = b_ih[2 * NH + jj];
  const float wim = fci_w[jj];
  const float bim = fci_b[0];

  // ---- t = 0 state: h = 0, gh = b_hh, cur = x[:,0] ----
  for (int e = tid; e < NB * NH; e += 256) h_s[e >> 7][e & 127] = 0.f;
  for (int e = tid; e < NB * NG; e += 256) gh_s[e / NG][e % NG] = b_hh[e % NG];
  if (tid < NB) {
    const float x0 = x_in[tid * NT];
    cur_s[tid] = x0;
    xin[tid * NT] = x0;
    pr_s[tid] = 0.f;
  }

  for (int t = 0; t < NT; ++t) {
    if (t > 0) {
      // A fragments of h (16x32 f16 tiles) from LDS
      v16h Af[4];
#pragma unroll
      for (int kb = 0; kb < 4; ++kb) {
        const _Float16* p = &h16_s[n][kb * 32 + hb * 8];
        v16h a;
#pragma unroll
        for (int i = 0; i < 8; ++i) { a[i] = p[i]; a[8 + i] = p[16 + i]; }
        Af[kb] = a;
      }
#pragma unroll
      for (int tt = 0; tt < 3; ++tt) {
        v8f acc;
#pragma unroll
        for (int r = 0; r < 8; ++r) acc[r] = binit[tt];   // fold b_hh into C
#pragma unroll
        for (int kb = 0; kb < 4; ++kb)
          acc = __builtin_amdgcn_wmma_f32_16x16x32_f16(
                    false, Af[kb], false, Bf[tt][kb], (short)0, acc, false, false);
        const int nt = wv * 3 + tt;
#pragma unroll
        for (int r = 0; r < 8; ++r)        // D: vgpr r -> M = r + hb*8, N = n
          gh_s[hb * 8 + r][nt * 16 + n] = acc[r];
      }
    }
    __syncthreads();

    // combine: h_t = GRUCell(cur_t, h_{t-1}); 8 (b,jj) elements per thread
#pragma unroll
    for (int i = 0; i < 8; ++i) {
      const int b = bb0 + 2 * i;
      const float cur  = cur_s[b];
      const float ghr  = gh_s[b][jj];
      const float ghz  = gh_s[b][NH + jj];
      const float ghn  = gh_s[b][2 * NH + jj];
      const float hold = h_s[b][jj];
      const float r  = fast_sigmoid(cur * wir + bir + ghr);
      const float z  = fast_sigmoid(cur * wiz + biz + ghz);
      const float nn = fast_tanh(cur * win + bin + r * ghn);
      const float hn = (1.f - z) * nn + z * hold;
      h_s[b][jj]   = hn;
      h16_s[b][jj] = (_Float16)hn;
      atomicAdd(&pr_s[b], hn * wim);       // ds_add_f32, prediction dot
    }
    __syncthreads();

    if (tid < NB) {
      const float p = pr_s[tid] + bim;     // = fc_imp(h_t) = x_hat_{t+1}
      if (t < NT - 1) {
        pred[tid * (NT - 1) + t] = p;
        const float xv = x_in[tid * NT + t + 1];
        const float c = (xv == SENT) ? p : xv;
        cur_s[tid] = c;
        xin[tid * NT + t + 1] = c;
      }
      pr_s[tid] = 0.f;
    }
    // next iteration's leading __syncthreads orders epilogue vs combine
  }
}

// ===========================================================================
// Kernel 2: layer-1 BN stats (sum, sumsq) per channel; y1 never materialized
// ===========================================================================
__global__ __launch_bounds__(256)
void k_stats1(const float* __restrict__ xin,
              const float* w0, const float* w1, const float* w2,
              const float* w3, const float* w4, const float* w5,
              float* __restrict__ stats1)
{
  __shared__ float xs[NPOS];
  __shared__ float red[512];
  const int tid = threadIdx.x;
  const float* wp[6] = {w0, w1, w2, w3, w4, w5};
  for (int i = tid; i < NPOS; i += 256) xs[i] = xin[i];
  __syncthreads();
  for (int ci = 0; ci < 12; ++ci) {
    const int c = blockIdx.x * 12 + ci;
    const int g = c / C1G, cc = c % C1G;
    const int k = prime_k(g);
    const float* w = wp[g] + cc * k;
    float s = 0.f, ss = 0.f;
    for (int e = tid; e < NPOS; e += 256) {
      const int b = e / NT, t = e - b * NT;
      const float y = conv1_eval(w, k, &xs[b * NT], t);
      s += y; ss += y * y;
    }
    blk_reduce2(s, ss, red, tid);
    if (tid == 0) { stats1[2 * c] = s; stats1[2 * c + 1] = ss; }
  }
}

// ===========================================================================
// Kernel 3: layer-1 BN affine (conv bias cancels inside BN: mean removal)
// ===========================================================================
__global__ __launch_bounds__(256)
void k_fin1(const float* __restrict__ stats1,
            const float* g0, const float* g1, const float* g2,
            const float* g3, const float* g4, const float* g5,
            const float* e0, const float* e1, const float* e2,
            const float* e3, const float* e4, const float* e5,
            float* __restrict__ a1, float* __restrict__ sh1)
{
  const int c = blockIdx.x * 256 + threadIdx.x;
  if (c >= NC1) return;
  const float* gm[6] = {g0, g1, g2, g3, g4, g5};
  const float* bt[6] = {e0, e1, e2, e3, e4, e5};
  const int g = c / C1G, cc = c % C1G;
  const float mean = stats1[2 * c] * (1.f / NPOS);
  const float var  = stats1[2 * c + 1] * (1.f / NPOS) - mean * mean;
  const float a = gm[g][cc] * rsqrtf(var + FEPS);
  a1[c] = a;
  sh1[c] = bt[g][cc] - mean * a;
}

__global__ void k_zero(float* __restrict__ y2) {
  const int i = blockIdx.x * 256 + threadIdx.x;
  if (i < 6 * NPOS) y2[i] = 0.f;
}

// ===========================================================================
// Kernel 4: layer-2 conv as channel reduction. Each block owns 32 input
// channels, regenerates BN+ReLU layer-1 rows, accumulates [6][16][300] in
// LDS, commits via global_atomic_add_f32.
// ===========================================================================
__global__ __launch_bounds__(256, 1)
void k_conv2(const float* __restrict__ xin,
             const float* __restrict__ a1, const float* __restrict__ sh1,
             const float* u0, const float* u1, const float* u2,
             const float* u3, const float* u4, const float* u5,   // layer-1 w
             const float* v0, const float* v1, const float* v2,
             const float* v3, const float* v4, const float* v5,   // layer-2 w
             float* __restrict__ y2)
{
  __shared__ float xs[NPOS];
  __shared__ float x2[NPOS];
  __shared__ float acc[6 * NPOS];     // 115.2 KB (gfx1250: 320 KB/WGP)
  const int tid = threadIdx.x;
  const float* w1p[6] = {u0, u1, u2, u3, u4, u5};
  const float* w2p[6] = {v0, v1, v2, v3, v4, v5};
  for (int i = tid; i < NPOS; i += 256) xs[i] = xin[i];
  for (int i = tid; i < 6 * NPOS; i += 256) acc[i] = 0.f;
  __syncthreads();

  const int c0 = blockIdx.x * 32;
  for (int ci = 0; ci < 32; ++ci) {
    const int c = c0 + ci;
    if (c < NC1) {
      const int g = c / C1G, cc = c % C1G;
      const int k = prime_k(g);
      const float* w = w1p[g] + cc * k;
      const float a = a1[c], sh = sh1[c];
      for (int e = tid; e < NPOS; e += 256) {
        const int b = e / NT, t = e - b * NT;
        x2[e] = fmaxf(a * conv1_eval(w, k, &xs[b * NT], t) + sh, 0.f);
      }
    }
    __syncthreads();
    if (c < NC1) {
      for (int e = tid; e < NPOS; e += 256) {
        const int b = e / NT, t = e - b * NT;
#pragma unroll
        for (int o = 0; o < 6; ++o) {
          const int ko = prime_k(o);
          const float* wr = w2p[o] + c * ko;       // (1, 9636, ko)
          const int half = (ko - 1) >> 1;
          float av = acc[o * NPOS + e];
          for (int j = 0; j < ko; ++j) {
            const int idx = t + (j - half) * 4;
            if (idx >= 0 && idx < NT) av += wr[j] * x2[b * NT + idx];
          }
          acc[o * NPOS + e] = av;
        }
      }
    }
    __syncthreads();
  }
  for (int i = tid; i < 6 * NPOS; i += 256) atomicAdd(&y2[i], acc[i]);
}

// ===========================================================================
// Kernel 5: layer-2 BN + ReLU -> x3[6][16][300]; also out[b] = fc_out.b
// ===========================================================================
__global__ __launch_bounds__(256)
void k_fin2(const float* __restrict__ y2,
            const float* g0, const float* g1, const float* g2,
            const float* g3, const float* g4, const float* g5,
            const float* e0, const float* e1, const float* e2,
            const float* e3, const float* e4, const float* e5,
            const float* __restrict__ fco_b,
            float* __restrict__ x3, float* __restrict__ out)
{
  __shared__ float red[512];
  __shared__ float ash[2];
  const int tid = threadIdx.x;
  const float* gm[6] = {g0, g1, g2, g3, g4, g5};
  const float* bt[6] = {e0, e1, e2, e3, e4, e5};
  for (int o = 0; o < 6; ++o) {
    float s = 0.f, ss = 0.f;
    for (int e = tid; e < NPOS; e += 256) { const float y = y2[o * NPOS + e]; s += y; ss += y * y; }
    blk_reduce2(s, ss, red, tid);
    if (tid == 0) {
      const float mean = s * (1.f / NPOS), var = ss * (1.f / NPOS) - mean * mean;
      const float a = gm[o][0] * rsqrtf(var + FEPS);
      ash[0] = a; ash[1] = bt[o][0] - mean * a;
    }
    __syncthreads();
    const float a = ash[0], sh = ash[1];
    for (int e = tid; e < NPOS; e += 256)
      x3[o * NPOS + e] = fmaxf(a * y2[o * NPOS + e] + sh, 0.f);
  }
  if (tid < NB) out[tid] = fco_b[0];
}

__device__ __forceinline__ void c3_map(int c, int& g, int& cc, int& k) {
  if (c < C3G)          { g = 0; cc = c; }
  else if (c < 2 * C3G) { g = 1; cc = c - C3G; }
  else                  { g = 2; cc = c - 2 * C3G; }
  k = 2 * g + 1;                                  // kernels 1, 3, 5
}

// ===========================================================================
// Kernel 6: layer-3 BN stats (y3 recomputed, never stored: saves 555 MB)
// ===========================================================================
__global__ __launch_bounds__(256, 1)
void k_stats3(const float* __restrict__ x3,
              const float* w0, const float* w1, const float* w2,
              float* __restrict__ stats3)
{
  __shared__ float x3s[6 * NPOS];
  __shared__ float red[512];
  const int tid = threadIdx.x;
  const float* wp[3] = {w0, w1, w2};
  for (int i = tid; i < 6 * NPOS; i += 256) x3s[i] = x3[i];
  __syncthreads();
  for (int ci = 0; ci < 12; ++ci) {
    const int c = blockIdx.x * 12 + ci;
    int g, cc, k; c3_map(c, g, cc, k);
    const float* w = wp[g] + cc * 6 * k;          // (9636, 6, k)
    const int half = (k - 1) >> 1;
    float s = 0.f, ss = 0.f;
    for (int e = tid; e < NPOS; e += 256) {
      const int b = e / NT, t = e - b * NT;
      float y = 0.f;
#pragma unroll
      for (int u = 0; u < 6; ++u) {
        const float* xr = &x3s[u * NPOS + b * NT];
        for (int j = 0; j < k; ++j) {
          const int idx = t + (j - half) * 4;
          if (idx >= 0 && idx < NT) y += w[u * k + j] * xr[idx];
        }
      }
      s += y; ss += y * y;
    }
    blk_reduce2(s, ss, red, tid);
    if (tid == 0) { stats3[2 * c] = s; stats3[2 * c + 1] = ss; }
  }
}

__global__ __launch_bounds__(256)
void k_fin3(const float* __restrict__ stats3,
            const float* g0, const float* g1, const float* g2,
            const float* e0, const float* e1, const float* e2,
            float* __restrict__ a3, float* __restrict__ sh3)
{
  const int c = blockIdx.x * 256 + threadIdx.x;
  if (c >= NC3) return;
  const float* gm[3] = {g0, g1, g2};
  const float* bt[3] = {e0, e1, e2};
  int g, cc, k; c3_map(c, g, cc, k);
  const float mean = stats3[2 * c] * (1.f / NPOS);
  const float var  = stats3[2 * c + 1] * (1.f / NPOS) - mean * mean;
  const float a = gm[g][cc] * rsqrtf(var + FEPS);
  a3[c] = a;
  sh3[c] = bt[g][cc] - mean * a;
}

// ===========================================================================
// Kernel 7: layer-3 recompute + BN + ReLU + GAP + fc_out, fused to out[b]
//   thread = b*16 + s owns time slots t = s, s+16, ... (fixed batch per thread)
// ===========================================================================
__global__ __launch_bounds__(256, 1)
void k_gap(const float* __restrict__ x3,
           const float* w0, const float* w1, const float* w2,
           const float* __restrict__ a3, const float* __restrict__ sh3,
           const float* __restrict__ fco_w,
           float* __restrict__ out)
{
  __shared__ float x3s[6 * NPOS];
  __shared__ float red[NB][17];
  const int tid = threadIdx.x;
  const int b = tid >> 4, sl = tid & 15;
  const float* wp[3] = {w0, w1, w2};
  for (int i = tid; i < 6 * NPOS; i += 256) x3s[i] = x3[i];
  __syncthreads();
  float accout = 0.f;                             // only meaningful for tid<16
  for (int ci = 0; ci < 12; ++ci) {
    const int c = blockIdx.x * 12 + ci;
    int g, cc, k; c3_map(c, g, cc, k);
    const float* w = wp[g] + cc * 6 * k;
    const int half = (k - 1) >> 1;
    const float a = a3[c], sh = sh3[c];
    float ps = 0.f;
    for (int t = sl; t < NT; t += 16) {
      float y = 0.f;
#pragma unroll
      for (int u = 0; u < 6; ++u) {
        const float* xr = &x3s[u * NPOS + b * NT];
        for (int j = 0; j < k; ++j) {
          const int idx = t + (j - half) * 4;
          if (idx >= 0 && idx < NT) y += w[u * k + j] * xr[idx];
        }
      }
      ps += fmaxf(a * y + sh, 0.f);
    }
    red[b][sl] = ps;
    __syncthreads();
    if (tid < NB) {
      float gs = 0.f;
#pragma unroll
      for (int i = 0; i < 16; ++i) gs += red[tid][i];
      accout += (gs * (1.f / NT)) * fco_w[c];
    }
    __syncthreads();
  }
  if (tid < NB) atomicAdd(&out[tid], accout);
}

// ===========================================================================
// Launch
// ===========================================================================
extern "C" void kernel_launch(void* const* d_in, const int* in_sizes, int n_in,
                              void* d_out, int out_size, void* d_ws, size_t ws_size,
                              hipStream_t stream)
{
  (void)n_in; (void)out_size; (void)ws_size;

  // ---- resolve input order: insertion order (x_input first, 4800 elems)
  //      vs jax sorted-key pytree flatten (conv b[1606] first).
  const bool ins = (in_sizes[0] == NPOS);
  const float *x_input, *g_wih, *g_whh, *g_bih, *g_bhh, *fci_w, *fci_b, *fco_w, *fco_b;
  const float *L1w[6], *L1g[6], *L1e[6];
  const float *L2w[6], *L2g[6], *L2e[6];
  const float *L3w[3], *L3g[3], *L3e[3];
  auto P = [&](int i) { return (const float*)d_in[i]; };
  if (ins) {
    x_input = P(0);
    g_wih = P(1); g_whh = P(2); g_bih = P(3); g_bhh = P(4);
    fci_w = P(5); fci_b = P(6);
    for (int i = 0; i < 6; ++i) { L1w[i]=P(7+4*i);  L1g[i]=P(9+4*i);  L1e[i]=P(10+4*i); }
    for (int i = 0; i < 6; ++i) { L2w[i]=P(31+4*i); L2g[i]=P(33+4*i); L2e[i]=P(34+4*i); }
    for (int i = 0; i < 3; ++i) { L3w[i]=P(55+4*i); L3g[i]=P(57+4*i); L3e[i]=P(58+4*i); }
    fco_w = P(67); fco_b = P(68);
  } else {     // sorted: conv(b,beta,gamma,w)*15, fc_imp(b,w), fc_out(b,w),
               //         gru(b_hh,b_ih,w_hh,w_ih), x_input
    for (int i = 0; i < 6; ++i) { L1e[i]=P(1+4*i);  L1g[i]=P(2+4*i);  L1w[i]=P(3+4*i); }
    for (int i = 0; i < 6; ++i) { L2e[i]=P(25+4*i); L2g[i]=P(26+4*i); L2w[i]=P(27+4*i); }
    for (int i = 0; i < 3; ++i) { L3e[i]=P(49+4*i); L3g[i]=P(50+4*i); L3w[i]=P(51+4*i); }
    fci_b = P(60); fci_w = P(61);
    fco_b = P(62); fco_w = P(63);
    g_bhh = P(64); g_bih = P(65); g_whh = P(66); g_wih = P(67);
    x_input = P(68);
  }

  float* out  = (float*)d_out;        // [16]
  float* pred = (float*)d_out + NB;   // [16*299]

  // ---- workspace layout (floats), ~846 KB total
  float* W      = (float*)d_ws;
  float* xin    = W;                  // 4800
  float* stats1 = xin + NPOS;         // 2*9636
  float* a1     = stats1 + 2 * NC1;   // 9636
  float* sh1    = a1 + NC1;           // 9636
  float* y2     = sh1 + NC1;          // 6*4800
  float* x3     = y2 + 6 * NPOS;      // 6*4800
  float* stats3 = x3 + 6 * NPOS;      // 2*28908
  float* a3     = stats3 + 2 * NC3;   // 28908
  float* sh3    = a3 + NC3;           // 28908

  k_zero<<<(6 * NPOS + 255) / 256, 256, 0, stream>>>(y2);
  k_gru<<<1, 256, 0, stream>>>(x_input, g_wih, g_whh, g_bih, g_bhh,
                               fci_w, fci_b, pred, xin);
  k_stats1<<<NC1 / 12, 256, 0, stream>>>(xin, L1w[0], L1w[1], L1w[2],
                                         L1w[3], L1w[4], L1w[5], stats1);
  k_fin1<<<(NC1 + 255) / 256, 256, 0, stream>>>(
      stats1, L1g[0], L1g[1], L1g[2], L1g[3], L1g[4], L1g[5],
      L1e[0], L1e[1], L1e[2], L1e[3], L1e[4], L1e[5], a1, sh1);
  k_conv2<<<(NC1 + 31) / 32, 256, 0, stream>>>(
      xin, a1, sh1,
      L1w[0], L1w[1], L1w[2], L1w[3], L1w[4], L1w[5],
      L2w[0], L2w[1], L2w[2], L2w[3], L2w[4], L2w[5], y2);
  k_fin2<<<1, 256, 0, stream>>>(
      y2, L2g[0], L2g[1], L2g[2], L2g[3], L2g[4], L2g[5],
      L2e[0], L2e[1], L2e[2], L2e[3], L2e[4], L2e[5], fco_b, x3, out);
  k_stats3<<<NC3 / 12, 256, 0, stream>>>(x3, L3w[0], L3w[1], L3w[2], stats3);
  k_fin3<<<(NC3 + 255) / 256, 256, 0, stream>>>(
      stats3, L3g[0], L3g[1], L3g[2], L3e[0], L3e[1], L3e[2], a3, sh3);
  k_gap<<<NC3 / 12, 256, 0, stream>>>(x3, L3w[0], L3w[1], L3w[2],
                                      a3, sh3, fco_w, out);
}
